// GATGraph_44590350467896
// MI455X (gfx1250) — matrix operand order repaired and checked by reading, not compiled, not run on hardware
//
#include <hip/hip_runtime.h>
#include <hip/hip_bf16.h>

typedef __attribute__((ext_vector_type(16))) _Float16 v16h;
typedef __attribute__((ext_vector_type(8)))  _Float16 v8h;
typedef __attribute__((ext_vector_type(8)))  float    v8f;

#define CH      128
#define OUTC    32
#define NGRAPH  64
#define NEGSLP  0.2f

// ---------- helpers: order-preserving float<->uint encoding for atomicMax ----------
__device__ __forceinline__ unsigned encf(float x) {
  unsigned u = __float_as_uint(x);
  return (u & 0x80000000u) ? ~u : (u | 0x80000000u);
}
__device__ __forceinline__ float decf(unsigned u) {
  return (u & 0x80000000u) ? __uint_as_float(u ^ 0x80000000u) : __uint_as_float(~u);
}
#define ENC_NEG_INF 0x007FFFFFu   // encf(-inf)

// ---------- elementwise utility kernels ----------
__global__ void k_cast_f16(const float* __restrict__ in, _Float16* __restrict__ out, int n) {
  int i = blockIdx.x * blockDim.x + threadIdx.x;
  if (i < n) out[i] = (_Float16)in[i];
}

__global__ void k_fill_u32(unsigned* __restrict__ p, unsigned v, int n) {
  int i = blockIdx.x * blockDim.x + threadIdx.x;
  if (i < n) p[i] = v;
}

// Pack 128x128 row-major weight W[k][n] into WMMA B-fragment order:
// Bp[colTile(8)][kstep(4)][lane(32)][elem(16)]
// lane holds column n = colTile*16 + (lane&15); K = kstep*32 + (lane>>4)*8 + (elem&7) + (elem>>3)*16
__global__ void k_pack_w(const float* __restrict__ W, _Float16* __restrict__ Bp) {
  int idx  = blockIdx.x * blockDim.x + threadIdx.x;  // 16384 total
  int e    = idx & 15;
  int lane = (idx >> 4) & 31;
  int ks   = (idx >> 9) & 3;
  int j    = idx >> 11;
  int n = j * 16 + (lane & 15);
  int k = ks * 32 + ((lane >> 4) * 8) + (e & 7) + ((e >> 3) << 4);
  Bp[idx] = (_Float16)W[k * CH + n];
}

// ---------- WMMA GEMM: Y[M x 128] = A_f16[M x 128] @ W  (W pre-packed) ----------
// block = (32,8): 8 waves, wave j owns output col-tile j; blockIdx.x = row tile (M/16)
__global__ void k_gemm_wmma(const _Float16* __restrict__ A,
                            const _Float16* __restrict__ Bp,
                            float* __restrict__ Y, int M) {
  int lane = threadIdx.x;          // 0..31
  int j    = threadIdx.y;          // 0..7 col tile
  int tm   = blockIdx.x;           // row tile
  int row  = tm * 16 + (lane & 15);
  int kb   = (lane >> 4) << 3;     // lanes 0-15 -> K 0..7/16..23 ; lanes 16-31 -> K 8..15/24..31
  v8f acc = {};
#pragma unroll
  for (int ks = 0; ks < 4; ++ks) {
    const _Float16* ap = A + (size_t)row * CH + ks * 32 + kb;
    v8h alo = *reinterpret_cast<const v8h*>(ap);        // K kb..kb+7
    v8h ahi = *reinterpret_cast<const v8h*>(ap + 16);   // K kb+16..kb+23
    v16h a;
#pragma unroll
    for (int i = 0; i < 8; ++i) { a[i] = alo[i]; a[8 + i] = ahi[i]; }
    v16h b = *reinterpret_cast<const v16h*>(Bp + ((((j * 4 + ks) * 32) + lane) << 4));
    acc = __builtin_amdgcn_wmma_f32_16x16x32_f16(false, a, false, b, (short)0, acc,
                                                 false, false);
  }
  int orow = tm * 16 + ((lane >> 4) << 3);   // C/D layout: vgpr r -> M = r (+8 for hi lanes)
  int ocol = j * 16 + (lane & 15);
#pragma unroll
  for (int r = 0; r < 8; ++r) Y[(size_t)(orow + r) * CH + ocol] = acc[r];
}

// ---------- edge phase ----------
// wave per edge: logit = sum_c lrelu(xr[dst][c]+xl[src][c]) * att[c]; atomic segment max
__global__ void k_edge_logits(const int* __restrict__ src, const int* __restrict__ dst,
                              const float* __restrict__ xl, const float* __restrict__ xr,
                              const float* __restrict__ att,
                              float* __restrict__ logits, unsigned* __restrict__ nmax, int E) {
  int lane = threadIdx.x & 31;
  int e = blockIdx.x * (blockDim.x >> 5) + (threadIdx.x >> 5);
  if (e >= E) return;
  int s = src[e], d = dst[e];
  float4 a  = reinterpret_cast<const float4*>(xl + (size_t)s * CH)[lane];
  float4 b  = reinterpret_cast<const float4*>(xr + (size_t)d * CH)[lane];
  float4 at = reinterpret_cast<const float4*>(att)[lane];
  float v0 = a.x + b.x, v1 = a.y + b.y, v2 = a.z + b.z, v3 = a.w + b.w;
  v0 = v0 > 0.f ? v0 : NEGSLP * v0;
  v1 = v1 > 0.f ? v1 : NEGSLP * v1;
  v2 = v2 > 0.f ? v2 : NEGSLP * v2;
  v3 = v3 > 0.f ? v3 : NEGSLP * v3;
  float sum = v0 * at.x + v1 * at.y + v2 * at.z + v3 * at.w;
#pragma unroll
  for (int off = 16; off > 0; off >>= 1) sum += __shfl_xor(sum, off, 32);
  if (lane == 0) {
    logits[e] = sum;
    atomicMax(&nmax[d], encf(sum));
  }
}

// thread per edge: ex = exp(logit - max[dst]); atomic segment sum
__global__ void k_edge_exp(const int* __restrict__ dst, float* __restrict__ logits,
                           const unsigned* __restrict__ nmax, float* __restrict__ nsum, int E) {
  int e = blockIdx.x * blockDim.x + threadIdx.x;
  if (e >= E) return;
  int d = dst[e];
  float ex = __expf(logits[e] - decf(nmax[d]));
  logits[e] = ex;
  atomicAdd(&nsum[d], ex);
}

// wave per edge: agg[dst] += alpha * xl[src]  (128 ch, 4 f32 atomics per lane)
__global__ void k_edge_agg(const int* __restrict__ src, const int* __restrict__ dst,
                           const float* __restrict__ xl, const float* __restrict__ logits,
                           const float* __restrict__ nsum, float* __restrict__ agg, int E) {
  int lane = threadIdx.x & 31;
  int e = blockIdx.x * (blockDim.x >> 5) + (threadIdx.x >> 5);
  if (e >= E) return;
  int s = src[e], d = dst[e];
  float alpha = logits[e] / nsum[d];
  float4 v = reinterpret_cast<const float4*>(xl + (size_t)s * CH)[lane];
  float* out = agg + (size_t)d * CH + lane * 4;
  atomicAdd(out + 0, alpha * v.x);
  atomicAdd(out + 1, alpha * v.y);
  atomicAdd(out + 2, alpha * v.z);
  atomicAdd(out + 3, alpha * v.w);
}

// h = relu(agg + bias); optionally also emit f16 copy for the next layer's GEMM
__global__ void k_node_finish(const float* __restrict__ agg, const float* __restrict__ bias,
                              float* __restrict__ h, _Float16* __restrict__ hf, int n) {
  int i = blockIdx.x * blockDim.x + threadIdx.x;
  if (i >= n) return;
  float v = agg[i] + bias[i & (CH - 1)];
  v = v > 0.f ? v : 0.f;
  h[i] = v;
  if (hf) hf[i] = (_Float16)v;
}

// ---------- pooling + final linear ----------
__global__ void k_pool(const float* __restrict__ h, const int* __restrict__ batch,
                       float* __restrict__ pooled, float* __restrict__ cnt, int N) {
  int lane = threadIdx.x & 31;
  int n = blockIdx.x * (blockDim.x >> 5) + (threadIdx.x >> 5);
  if (n >= N) return;
  int g = batch[n];
  float4 v = reinterpret_cast<const float4*>(h + (size_t)n * CH)[lane];
  float* out = pooled + (size_t)g * CH + lane * 4;
  atomicAdd(out + 0, v.x);
  atomicAdd(out + 1, v.y);
  atomicAdd(out + 2, v.z);
  atomicAdd(out + 3, v.w);
  if (lane == 0) atomicAdd(&cnt[g], 1.0f);
}

__global__ void k_final(const float* __restrict__ pooled, const float* __restrict__ cnt,
                        const float* __restrict__ linW, const float* __restrict__ linb,
                        float* __restrict__ out) {
  int g = blockIdx.x;   // 0..63
  int o = threadIdx.x;  // 0..31
  float inv = 1.0f / fmaxf(cnt[g], 1.0f);
  float s = 0.f;
#pragma unroll 4
  for (int c = 0; c < CH; ++c) s += pooled[g * CH + c] * linW[c * OUTC + o];
  out[g * OUTC + o] = s * inv + linb[o];
}

// ---------- driver ----------
extern "C" void kernel_launch(void* const* d_in, const int* in_sizes, int n_in,
                              void* d_out, int out_size, void* d_ws, size_t ws_size,
                              hipStream_t stream) {
  const float* x     = (const float*)d_in[0];
  const int*   ei    = (const int*)  d_in[1];
  const int*   batch = (const int*)  d_in[2];
  const float* Wl0   = (const float*)d_in[3];
  const float* Wr0   = (const float*)d_in[4];
  const float* att0  = (const float*)d_in[5];
  const float* b0    = (const float*)d_in[6];
  const float* Wl1   = (const float*)d_in[7];
  const float* Wr1   = (const float*)d_in[8];
  const float* att1  = (const float*)d_in[9];
  const float* b1    = (const float*)d_in[10];
  const float* linW  = (const float*)d_in[11];
  const float* linb  = (const float*)d_in[12];

  const int N = in_sizes[0] / CH;
  const int E = in_sizes[1] / 2;
  const int* src = ei;
  const int* dst = ei + E;
  const int nNodeF = N * CH;

  // carve workspace
  char* p = (char*)d_ws;
  auto take = [&](size_t bytes) -> void* {
    void* r = (void*)p;
    p += (bytes + 255) & ~(size_t)255;
    return r;
  };
  float*    xl     = (float*)   take((size_t)nNodeF * 4);
  float*    xr     = (float*)   take((size_t)nNodeF * 4);
  float*    h0     = (float*)   take((size_t)nNodeF * 4);   // agg0 -> h0 (in place)
  float*    h1     = (float*)   take((size_t)nNodeF * 4);   // agg1 -> h1 (in place)
  _Float16* hf     = (_Float16*)take((size_t)nNodeF * 2);
  _Float16* wlp    = (_Float16*)take((size_t)16384 * 2);
  _Float16* wrp    = (_Float16*)take((size_t)16384 * 2);
  float*    lg     = (float*)   take((size_t)E * 4);
  unsigned* nmax   = (unsigned*)take((size_t)N * 4);
  float*    nsum   = (float*)   take((size_t)N * 4);
  float*    pooled = (float*)   take((size_t)NGRAPH * CH * 4);
  float*    cnt    = (float*)   take((size_t)NGRAPH * 4);

  dim3 b256(256);
  dim3 bgemm(32, 8);
  int gNodeF = (nNodeF + 255) / 256;
  int gN     = (N + 255) / 256;
  int gE1    = (E + 255) / 256;
  int gE8    = (E + 7) / 8;       // 8 waves/block, wave per edge
  int gM     = N / 16;            // N = 50000 = 3125*16

  // ---- layer 0 ----
  k_cast_f16<<<gNodeF, b256, 0, stream>>>(x, hf, nNodeF);
  k_pack_w<<<64, b256, 0, stream>>>(Wl0, wlp);
  k_pack_w<<<64, b256, 0, stream>>>(Wr0, wrp);
  k_gemm_wmma<<<gM, bgemm, 0, stream>>>(hf, wlp, xl, N);
  k_gemm_wmma<<<gM, bgemm, 0, stream>>>(hf, wrp, xr, N);
  k_fill_u32<<<gN, b256, 0, stream>>>(nmax, ENC_NEG_INF, N);
  k_fill_u32<<<gN, b256, 0, stream>>>((unsigned*)nsum, 0u, N);
  k_edge_logits<<<gE8, b256, 0, stream>>>(src, dst, xl, xr, att0, lg, nmax, E);
  k_edge_exp<<<gE1, b256, 0, stream>>>(dst, lg, nmax, nsum, E);
  k_fill_u32<<<gNodeF, b256, 0, stream>>>((unsigned*)h0, 0u, nNodeF);
  k_edge_agg<<<gE8, b256, 0, stream>>>(src, dst, xl, lg, nsum, h0, E);
  k_node_finish<<<gNodeF, b256, 0, stream>>>(h0, b0, h0, hf, nNodeF);

  // ---- layer 1 ----
  k_pack_w<<<64, b256, 0, stream>>>(Wl1, wlp);
  k_pack_w<<<64, b256, 0, stream>>>(Wr1, wrp);
  k_gemm_wmma<<<gM, bgemm, 0, stream>>>(hf, wlp, xl, N);
  k_gemm_wmma<<<gM, bgemm, 0, stream>>>(hf, wrp, xr, N);
  k_fill_u32<<<gN, b256, 0, stream>>>(nmax, ENC_NEG_INF, N);
  k_fill_u32<<<gN, b256, 0, stream>>>((unsigned*)nsum, 0u, N);
  k_edge_logits<<<gE8, b256, 0, stream>>>(src, dst, xl, xr, att1, lg, nmax, E);
  k_edge_exp<<<gE1, b256, 0, stream>>>(dst, lg, nmax, nsum, E);
  k_fill_u32<<<gNodeF, b256, 0, stream>>>((unsigned*)h1, 0u, nNodeF);
  k_edge_agg<<<gE8, b256, 0, stream>>>(src, dst, xl, lg, nsum, h1, E);
  k_node_finish<<<gNodeF, b256, 0, stream>>>(h1, b1, h1, (_Float16*)nullptr, nNodeF);

  // ---- pool + linear ----
  k_fill_u32<<<(NGRAPH * CH + 255) / 256, b256, 0, stream>>>((unsigned*)pooled, 0u, NGRAPH * CH);
  k_fill_u32<<<1, b256, 0, stream>>>((unsigned*)cnt, 0u, NGRAPH);
  k_pool<<<(N + 7) / 8, b256, 0, stream>>>(h1, batch, pooled, cnt, N);
  k_final<<<NGRAPH, OUTC, 0, stream>>>(pooled, cnt, linW, linb, (float*)d_out);
}
